// GPT_30588757082713
// MI455X (gfx1250) — matrix operand order repaired and checked by reading, not compiled, not run on hardware
//
#include <hip/hip_runtime.h>
#include <math.h>

#define Bsz 2
#define Ssz 1024
#define Dsz 768
#define Hn  12
#define Ln  4
#define Vsz 32000
#define DHsz 64
#define Msz (Bsz*Ssz)      // 2048
#define Fsz (4*Dsz)        // 3072

typedef __attribute__((ext_vector_type(16))) __bf16 v16bf;
typedef __attribute__((ext_vector_type(8)))  float  v8f;

struct Frag32B { uint4 lo, hi; };  // 32 bytes == v16bf

// Native fptrunc: gfx1250 has hardware BF16 converts (v_cvt_pk_bf16_f32 class),
// far cheaper than the manual bfe/add3 RNE sequence.
static __device__ __forceinline__ unsigned short f2bf(float f) {
    __bf16 b = (__bf16)f;
    return __builtin_bit_cast(unsigned short, b);
}

static __device__ __forceinline__ v16bf load_frag(const unsigned short* plo,
                                                  const unsigned short* phi) {
    Frag32B f;
    f.lo = *(const uint4*)plo;
    f.hi = *(const uint4*)phi;
    return __builtin_bit_cast(v16bf, f);
}

// ---------------------------------------------------------------------------
// Embedding: x = tok_emb[tok] * sqrt(D) + PE(s, d)
// ---------------------------------------------------------------------------
__global__ __launch_bounds__(256) void embed_kernel(const int* __restrict__ tokens,
                                                    const float* __restrict__ emb,
                                                    float* __restrict__ X) {
    const int row = blockIdx.x;            // b*S + s
    const int s   = row & (Ssz - 1);
    const int tok = tokens[row];
    const float* e = emb + (size_t)tok * Dsz;
    float* xr = X + (size_t)row * Dsz;
    for (int d = threadIdx.x; d < Dsz; d += 256) {
        int i = d >> 1;
        float freq = __expf(-9.210340371976184f * (float)(2 * i) / (float)Dsz);
        float ang  = (float)s * freq;
        float pe   = (d & 1) ? cosf(ang) : sinf(ang);
        xr[d] = e[d] * 27.712812921102035f + pe;   // sqrt(768)
    }
}

// ---------------------------------------------------------------------------
// LayerNorm (row = one block of 256 threads, D = 768)
// ---------------------------------------------------------------------------
__global__ __launch_bounds__(256) void ln_kernel(const float* __restrict__ X,
                                                 const float* __restrict__ g,
                                                 const float* __restrict__ bta,
                                                 float* __restrict__ O) {
    __shared__ float red[16];
    const int row = blockIdx.x, tid = threadIdx.x;
    const float* xr = X + (size_t)row * Dsz;
    float s = 0.f, s2 = 0.f;
    for (int d = tid; d < Dsz; d += 256) { float v = xr[d]; s += v; s2 += v * v; }
#pragma unroll
    for (int off = 16; off >= 1; off >>= 1) {
        s  += __shfl_xor(s,  off, 32);
        s2 += __shfl_xor(s2, off, 32);
    }
    if ((tid & 31) == 0) { red[(tid >> 5) * 2] = s; red[(tid >> 5) * 2 + 1] = s2; }
    __syncthreads();
    s = 0.f; s2 = 0.f;
#pragma unroll
    for (int i = 0; i < 8; i++) { s += red[i * 2]; s2 += red[i * 2 + 1]; }
    const float mu   = s * (1.0f / Dsz);
    const float var  = s2 * (1.0f / Dsz) - mu * mu;
    const float rstd = rsqrtf(var + 1e-5f);
    float* orow = O + (size_t)row * Dsz;
    for (int d = tid; d < Dsz; d += 256)
        orow[d] = (xr[d] - mu) * rstd * g[d] + bta[d];
}

// ---------------------------------------------------------------------------
// GEMM body: C[M,N] = act(A[M,K] @ B[K,N] + bias) (+ Res)    bf16 WMMA, fp32 I/O
// tile 128x128x32, 256 threads = 8 waves, each wave 64x32 (4x2 wmma tiles).
// Double-buffered LDS + register prefetch: one barrier per k-step, next tile's
// global loads issue before the WMMA block so HBM/L2 latency overlaps compute.
// ---------------------------------------------------------------------------
template<bool GELU, bool RES>
static __device__ __forceinline__ void gemm_body(const float* __restrict__ A,
                                                 const float* __restrict__ Bw,
                                                 const float* __restrict__ bias,
                                                 const float* __restrict__ Res,
                                                 float* __restrict__ C,
                                                 int M, int N, int K,
                                                 int m0, int n0) {
    constexpr int ASld = 40;    // shorts per A row (32 + pad), 80B: 16B-aligned frags
    constexpr int BSld = 136;   // shorts per B row (128 + pad), 272B
    __shared__ __align__(16) unsigned short As[2][128 * ASld];
    __shared__ __align__(16) unsigned short Bs[2][32 * BSld];

    const int tid  = threadIdx.x;
    const int lane = tid & 31, w = tid >> 5;
    const int wrow = (w >> 2) * 64;     // wave M offset in tile
    const int wcol = (w & 3) * 32;      // wave N offset in tile
    const int ml = lane & 15, kb = (lane >> 4) * 8;

    float4 ra[4], rb[4];
    auto gload = [&](int k0) {
#pragma unroll
        for (int i = 0; i < 4; i++) {
            int idx = tid + i * 256;
            ra[i] = *(const float4*)(A + (size_t)(m0 + (idx >> 3)) * K + k0 + (idx & 7) * 4);
            rb[i] = *(const float4*)(Bw + (size_t)(k0 + (idx >> 5)) * N + n0 + (idx & 31) * 4);
        }
    };
    auto stage = [&](int p) {
#pragma unroll
        for (int i = 0; i < 4; i++) {
            int idx = tid + i * 256;
            ushort4 ha, hb;
            ha.x = f2bf(ra[i].x); ha.y = f2bf(ra[i].y); ha.z = f2bf(ra[i].z); ha.w = f2bf(ra[i].w);
            hb.x = f2bf(rb[i].x); hb.y = f2bf(rb[i].y); hb.z = f2bf(rb[i].z); hb.w = f2bf(rb[i].w);
            *(ushort4*)&As[p][(idx >> 3) * ASld + (idx & 7) * 4] = ha;
            *(ushort4*)&Bs[p][(idx >> 5) * BSld + (idx & 31) * 4] = hb;
        }
    };

    v8f acc[4][2];
#pragma unroll
    for (int mt = 0; mt < 4; mt++)
#pragma unroll
        for (int nt = 0; nt < 2; nt++) acc[mt][nt] = (v8f)0.0f;

    gload(0);
    int p = 0;
    for (int k0 = 0; k0 < K; k0 += 32) {
        stage(p);
        __syncthreads();
        if (k0 + 32 < K) {
            gload(k0 + 32);                    // overlap next tile with WMMA below
            if (k0 + 64 < K)                   // L2 prefetch two tiles ahead
                __builtin_prefetch(Bw + (size_t)(k0 + 64 + (tid >> 5)) * N + n0 + (tid & 31) * 4, 0, 1);
        }
        v16bf a[4];
#pragma unroll
        for (int mt = 0; mt < 4; mt++) {
            const unsigned short* rp = &As[p][(wrow + mt * 16 + ml) * ASld];
            a[mt] = load_frag(rp + kb, rp + 16 + kb);
        }
#pragma unroll
        for (int nt = 0; nt < 2; nt++) {
            const unsigned short* rp = &Bs[p][kb * BSld + wcol + nt * 16 + ml];
            v16bf b;
#pragma unroll
            for (int s = 0; s < 8; s++) {
                b[s]     = __builtin_bit_cast(__bf16, rp[s * BSld]);
                b[8 + s] = __builtin_bit_cast(__bf16, rp[(16 + s) * BSld]);
            }
#pragma unroll
            for (int mt = 0; mt < 4; mt++)
                acc[mt][nt] = __builtin_amdgcn_wmma_f32_16x16x32_bf16(
                    false, a[mt], false, b, (short)0, acc[mt][nt], false, false);
        }
        p ^= 1;    // buffer p rewritten only after a full barrier separates its readers
    }

    // epilogue: bias (+gelu) (+residual)
#pragma unroll
    for (int nt = 0; nt < 2; nt++) {
        const int n = n0 + wcol + nt * 16 + ml;
        const float bv = bias[n];
#pragma unroll
        for (int mt = 0; mt < 4; mt++) {
#pragma unroll
            for (int r = 0; r < 8; r++) {
                float v = acc[mt][nt][r] + bv;
                if (GELU) v = 0.5f * v * (1.0f + erff(v * 0.7071067811865475f));
                const size_t off = (size_t)(m0 + wrow + mt * 16 + kb + r) * N + n;
                if (RES) v += Res[off];
                C[off] = v;
            }
        }
    }
}

template<bool GELU, bool RES>
__global__ __launch_bounds__(256) void gemm_kernel(const float* __restrict__ A,
                                                   const float* __restrict__ Bw,
                                                   const float* __restrict__ bias,
                                                   const float* __restrict__ Res,
                                                   float* __restrict__ C,
                                                   int M, int N, int K) {
    gemm_body<GELU, RES>(A, Bw, bias, Res, C, M, N, K, blockIdx.y * 128, blockIdx.x * 128);
}

// Fused QKV: one launch, grid.x = 3 * (D/128); blockIdx.x/6 selects {Wq,Wk,Wv}
__global__ __launch_bounds__(256) void qkv_kernel(const float* __restrict__ h,
                                                  const float* __restrict__ Wq,
                                                  const float* __restrict__ Wk,
                                                  const float* __restrict__ Wv,
                                                  const float* __restrict__ bq,
                                                  const float* __restrict__ bk,
                                                  const float* __restrict__ bv,
                                                  float* __restrict__ q,
                                                  float* __restrict__ k,
                                                  float* __restrict__ v) {
    const int sel = blockIdx.x / (Dsz / 128);
    const int n0  = (blockIdx.x % (Dsz / 128)) * 128;
    const float* Bw   = (sel == 0) ? Wq : (sel == 1) ? Wk : Wv;
    const float* bias = (sel == 0) ? bq : (sel == 1) ? bk : bv;
    float*       C    = (sel == 0) ? q  : (sel == 1) ? k  : v;
    gemm_body<false, false>(h, Bw, bias, nullptr, C, Msz, Dsz, Dsz, blockIdx.y * 128, n0);
}

// ---------------------------------------------------------------------------
// Flash attention: 1 WG = 64 q rows of one (b,h); 4 waves x 16 rows.
// Q*K^T and P*V via WMMA bf16; online softmax; causal mask = -1e6.
// ---------------------------------------------------------------------------
__global__ __launch_bounds__(128) void attn_kernel(const float* __restrict__ Qg,
                                                   const float* __restrict__ Kg,
                                                   const float* __restrict__ Vg,
                                                   float* __restrict__ Y) {
    constexpr int ST = 72;          // row stride in shorts (144B, 16B multiple)
    __shared__ __align__(16) unsigned short Qs[64 * ST];     // [q][d]   * 0.125
    __shared__ __align__(16) unsigned short Ks[64 * ST];     // [d][kv]  (transposed!)
    __shared__ __align__(16) unsigned short Vs[64 * ST];     // [kv][d]
    __shared__ __align__(16) unsigned short Ps[4 * 16 * ST]; // per-wave P tile

    const int tid = threadIdx.x, lane = tid & 31, w = tid >> 5;
    const int qblk = blockIdx.x, q0 = qblk * 64;
    const int bh = blockIdx.y, b = bh / Hn, h = bh % Hn;
    const size_t basebh = ((size_t)b * Ssz) * Dsz + (size_t)h * DHsz;
    const int ml = lane & 15, kb = (lane >> 4) * 8;

    // stage Q (scaled by 1/sqrt(DH) = 0.125)
#pragma unroll
    for (int i = 0; i < 8; i++) {
        int idx = tid + i * 128;
        int r = idx >> 4, c4 = idx & 15;
        float4 f = *(const float4*)(Qg + basebh + (size_t)(q0 + r) * Dsz + c4 * 4);
        ushort4 hc;
        hc.x = f2bf(f.x * 0.125f); hc.y = f2bf(f.y * 0.125f);
        hc.z = f2bf(f.z * 0.125f); hc.w = f2bf(f.w * 0.125f);
        *(ushort4*)&Qs[r * ST + c4 * 4] = hc;
    }
    __syncthreads();

    v16bf qa[2];
#pragma unroll
    for (int c = 0; c < 2; c++) {
        const unsigned short* rp = &Qs[(w * 16 + ml) * ST + c * 32];
        qa[c] = load_frag(rp + kb, rp + 16 + kb);
    }

    v8f o[4];
    float mi[8], li[8];
#pragma unroll
    for (int nt = 0; nt < 4; nt++) o[nt] = (v8f)0.0f;
#pragma unroll
    for (int r = 0; r < 8; r++) { mi[r] = -1e30f; li[r] = 0.0f; }

    const int nblk = qblk + 1;                 // causal: kv blocks 0..qblk
    for (int j = 0; j < nblk; j++) {
        __syncthreads();                        // protect Ks/Vs reuse
        const int kv0 = j * 64;
#pragma unroll
        for (int i = 0; i < 8; i++) {
            int idx = tid + i * 128;
            int r = idx >> 4, c4 = idx & 15;    // r = kv row, c4*4 = d
            float4 fk = *(const float4*)(Kg + basebh + (size_t)(kv0 + r) * Dsz + c4 * 4);
            Ks[(c4 * 4 + 0) * ST + r] = f2bf(fk.x);
            Ks[(c4 * 4 + 1) * ST + r] = f2bf(fk.y);
            Ks[(c4 * 4 + 2) * ST + r] = f2bf(fk.z);
            Ks[(c4 * 4 + 3) * ST + r] = f2bf(fk.w);
            float4 fv = *(const float4*)(Vg + basebh + (size_t)(kv0 + r) * Dsz + c4 * 4);
            ushort4 hv; hv.x = f2bf(fv.x); hv.y = f2bf(fv.y); hv.z = f2bf(fv.z); hv.w = f2bf(fv.w);
            *(ushort4*)&Vs[r * ST + c4 * 4] = hv;
        }
        __syncthreads();

        // scores: S = Q*K^T  (4 kv tiles of 16, K-dim 64 = 2 chunks)
        v8f sc[4];
#pragma unroll
        for (int nt = 0; nt < 4; nt++) {
            sc[nt] = (v8f)0.0f;
#pragma unroll
            for (int c = 0; c < 2; c++) {
                const unsigned short* rp = &Ks[(c * 32 + kb) * ST + nt * 16 + ml];
                v16bf kbf;
#pragma unroll
                for (int s = 0; s < 8; s++) {
                    kbf[s]     = __builtin_bit_cast(__bf16, rp[s * ST]);
                    kbf[8 + s] = __builtin_bit_cast(__bf16, rp[(16 + s) * ST]);
                }
                sc[nt] = __builtin_amdgcn_wmma_f32_16x16x32_bf16(
                    false, qa[c], false, kbf, (short)0, sc[nt], false, false);
            }
        }
        if (j == qblk) {                         // diagonal block: causal mask
#pragma unroll
            for (int nt = 0; nt < 4; nt++) {
                const int ng = kv0 + nt * 16 + ml;
#pragma unroll
                for (int r = 0; r < 8; r++) {
                    const int mg = q0 + w * 16 + r + 8 * (lane >> 4);
                    if (ng > mg) sc[nt][r] = -1e6f;
                }
            }
        }

        // online softmax
        float alpha[8];
#pragma unroll
        for (int r = 0; r < 8; r++) {
            float mx = fmaxf(fmaxf(sc[0][r], sc[1][r]), fmaxf(sc[2][r], sc[3][r]));
#pragma unroll
            for (int off = 8; off >= 1; off >>= 1) mx = fmaxf(mx, __shfl_xor(mx, off, 32));
            const float mnew = fmaxf(mi[r], mx);
            alpha[r] = __expf(mi[r] - mnew);
            float rsum = 0.0f;
#pragma unroll
            for (int nt = 0; nt < 4; nt++) {
                float ppv = __expf(sc[nt][r] - mnew);
                sc[nt][r] = ppv;
                rsum += ppv;
            }
#pragma unroll
            for (int off = 8; off >= 1; off >>= 1) rsum += __shfl_xor(rsum, off, 32);
            li[r] = li[r] * alpha[r] + rsum;
            mi[r] = mnew;
        }

        // P -> LDS (re-layout D-fragment -> A-fragment)
        unsigned short* pw = &Ps[w * 16 * ST];
#pragma unroll
        for (int nt = 0; nt < 4; nt++)
#pragma unroll
            for (int r = 0; r < 8; r++)
                pw[(r + 8 * (lane >> 4)) * ST + nt * 16 + ml] = f2bf(sc[nt][r]);
        __syncthreads();

#pragma unroll
        for (int nt = 0; nt < 4; nt++)
#pragma unroll
            for (int r = 0; r < 8; r++) o[nt][r] *= alpha[r];

        // O += P @ V
#pragma unroll
        for (int c = 0; c < 2; c++) {
            const unsigned short* rp = &pw[ml * ST + c * 32];
            v16bf pa = load_frag(rp + kb, rp + 16 + kb);
#pragma unroll
            for (int nt = 0; nt < 4; nt++) {
                const unsigned short* vp = &Vs[(c * 32 + kb) * ST + nt * 16 + ml];
                v16bf vb;
#pragma unroll
                for (int s = 0; s < 8; s++) {
                    vb[s]     = __builtin_bit_cast(__bf16, vp[s * ST]);
                    vb[8 + s] = __builtin_bit_cast(__bf16, vp[(16 + s) * ST]);
                }
                o[nt] = __builtin_amdgcn_wmma_f32_16x16x32_bf16(
                    false, pa, false, vb, (short)0, o[nt], false, false);
            }
        }
    }

    // normalize & store
#pragma unroll
    for (int r = 0; r < 8; r++) {
        const float inv = 1.0f / li[r];
        const int sg = q0 + w * 16 + r + 8 * (lane >> 4);
        const size_t rowo = basebh + (size_t)sg * Dsz;
#pragma unroll
        for (int nt = 0; nt < 4; nt++)
            Y[rowo + nt * 16 + ml] = o[nt][r] * inv;
    }
}

// ---------------------------------------------------------------------------
extern "C" void kernel_launch(void* const* d_in, const int* in_sizes, int n_in,
                              void* d_out, int out_size, void* d_ws, size_t ws_size,
                              hipStream_t stream) {
    (void)in_sizes; (void)n_in; (void)out_size; (void)ws_size;
    const int*   tokens = (const int*)  d_in[0];
    const float* tokemb = (const float*)d_in[1];
    const float* Wq = (const float*)d_in[2],  *bq = (const float*)d_in[3];
    const float* Wk = (const float*)d_in[4],  *bk = (const float*)d_in[5];
    const float* Wv = (const float*)d_in[6],  *bv = (const float*)d_in[7];
    const float* Wo = (const float*)d_in[8],  *bo = (const float*)d_in[9];
    const float* g1 = (const float*)d_in[10], *be1 = (const float*)d_in[11];
    const float* g2 = (const float*)d_in[12], *be2 = (const float*)d_in[13];
    const float* W1 = (const float*)d_in[14], *b1 = (const float*)d_in[15];
    const float* W2 = (const float*)d_in[16], *b2 = (const float*)d_in[17];
    const float* lnfg = (const float*)d_in[18], *lnfb = (const float*)d_in[19];
    const float* fcW = (const float*)d_in[20], *fcb = (const float*)d_in[21];
    float* out = (float*)d_out;

    float* ws = (float*)d_ws;
    const size_t MD = (size_t)Msz * Dsz;
    float* x  = ws;            // [M,D] residual stream
    float* h  = x  + MD;       // [M,D] LN output / GEMM A
    float* q  = h  + MD;
    float* k  = q  + MD;
    float* v  = k  + MD;
    float* y  = v  + MD;       // attention output
    float* ff = y  + MD;       // [M,4D]

    const dim3 blk256(256), blk128(128);
    const dim3 gD(Dsz / 128, Msz / 128);       // N=768
    const dim3 gQKV(3 * Dsz / 128, Msz / 128); // fused q,k,v
    const dim3 gF(Fsz / 128, Msz / 128);       // N=3072
    const dim3 gV(Vsz / 128, Msz / 128);       // N=32000
    const dim3 gAtt(Ssz / 64, Bsz * Hn);

    embed_kernel<<<Msz, blk256, 0, stream>>>(tokens, tokemb, x);

    for (int l = 0; l < Ln; l++) {
        const size_t wofs = (size_t)l * Dsz * Dsz;
        const size_t fofs = (size_t)l * Dsz * Fsz;
        ln_kernel<<<Msz, blk256, 0, stream>>>(x, g1 + l * Dsz, be1 + l * Dsz, h);
        qkv_kernel<<<gQKV, blk256, 0, stream>>>(h, Wq + wofs, Wk + wofs, Wv + wofs,
                                                bq + l * Dsz, bk + l * Dsz, bv + l * Dsz,
                                                q, k, v);
        attn_kernel<<<gAtt, blk128, 0, stream>>>(q, k, v, y);
        gemm_kernel<false, true ><<<gD, blk256, 0, stream>>>(y, Wo + wofs, bo + l * Dsz, x, x, Msz, Dsz, Dsz);
        ln_kernel<<<Msz, blk256, 0, stream>>>(x, g2 + l * Dsz, be2 + l * Dsz, h);
        gemm_kernel<true,  false><<<gF, blk256, 0, stream>>>(h, W1 + fofs, b1 + (size_t)l * Fsz, nullptr, ff, Msz, Fsz, Dsz);
        gemm_kernel<false, true ><<<gD, blk256, 0, stream>>>(ff, W2 + fofs, b2 + l * Dsz, x, x, Msz, Dsz, Fsz);
    }

    ln_kernel<<<Msz, blk256, 0, stream>>>(x, lnfg, lnfb, h);
    gemm_kernel<false, false><<<gV, blk256, 0, stream>>>(h, fcW, fcb, nullptr, out, Msz, Vsz, Dsz);
}
